// Model_5909874999523
// MI455X (gfx1250) — matrix operand order repaired and checked by reading, not compiled, not run on hardware
//
#include <hip/hip_runtime.h>
#include <hip/hip_bf16.h>
#include <math.h>

typedef __attribute__((ext_vector_type(16))) _Float16 v16h;
typedef __attribute__((ext_vector_type(8)))  _Float16 v8h;
typedef __attribute__((ext_vector_type(8)))  float    v8f;
typedef __attribute__((ext_vector_type(4)))  float    v4f;

#define NB 2
#define NH 16
#define NS 2048
#define ND 64
#define KTILE 32
#define WAVES 4
#define QB 64          // WAVES * 16 query rows per block
#define NKT (NS / KTILE)

__device__ __forceinline__ v16h cat16(v8h lo, v8h hi) {
  v16h r;
#pragma unroll
  for (int i = 0; i < 8; ++i) { r[i] = lo[i]; r[i + 8] = hi[i]; }
  return r;
}

// load 8 consecutive f32 from global, convert to 8 f16
__device__ __forceinline__ v8h cvt8(const float* p) {
  v4f a = *(const v4f*)p;
  v4f b = *(const v4f*)(p + 4);
  v8h r;
#pragma unroll
  for (int i = 0; i < 4; ++i) { r[i] = (_Float16)a[i]; r[i + 4] = (_Float16)b[i]; }
  return r;
}

// ---- DPP16 cross-lane moves (VALU, no LDS round-trip) ----
// 0xB1 = quad_perm(1,0,3,2) == xor1 ; 0x4E = quad_perm(2,3,0,1) == xor2
// 0x141 = row_half_mirror (lane^7 in 8-group) ; 0x140 = row_mirror (lane^15 in 16)
template <int CTRL>
__device__ __forceinline__ float dppmov(float x) {
  return __int_as_float(__builtin_amdgcn_update_dpp(
      0, __float_as_int(x), CTRL, 0xf, 0xf, true));
}
// full reduction over each 16-lane half, result replicated to all lanes
__device__ __forceinline__ float red16_max(float t) {
  t = fmaxf(t, dppmov<0xB1>(t));
  t = fmaxf(t, dppmov<0x4E>(t));
  t = fmaxf(t, dppmov<0x141>(t));
  t = fmaxf(t, dppmov<0x140>(t));
  return t;
}
__device__ __forceinline__ float red16_sum(float t) {
  t += dppmov<0xB1>(t);
  t += dppmov<0x4E>(t);
  t += dppmov<0x141>(t);
  t += dppmov<0x140>(t);
  return t;
}

__global__ void __launch_bounds__(128)
fattn_wmma_f16(const float* __restrict__ Qg_, const float* __restrict__ Kg_,
               const float* __restrict__ Vg_, const float* __restrict__ Mg,
               const float* __restrict__ Dg_, float* __restrict__ Og_) {
  // Double-buffered K (row-major [k][d]) and V (transposed [d][k]) f16 tiles,
  // plus per-wave P staging. Row strides padded to multiples of 16B.
  __shared__ __align__(16) _Float16 Ks[2][KTILE][72];     // 144B stride
  __shared__ __align__(16) _Float16 Vt[2][ND][40];        // 80B stride
  __shared__ __align__(16) _Float16 Ps[WAVES][16][40];    // 80B stride

  const int tid  = threadIdx.x;
  const int wave = tid >> 5;
  const int lane = tid & 31;
  const int bh   = blockIdx.y;                 // b*H + h
  const int qbase = blockIdx.x * QB + wave * 16;

  const size_t bhSD = (size_t)bh * NS * ND;
  const float* Qg = Qg_ + bhSD;
  const float* Kg = Kg_ + bhSD;
  const float* Vg = Vg_ + bhSD;
  const float* Dg = Dg_ + (size_t)bh * NS * NS;
  float*       Og = Og_ + bhSD;

  // A-fragment coords (16-bit A 16x32): lane<16 -> M=lane, K in {0..15}
  const int arow  = lane & 15;
  const int abase = (lane >> 4) * 16;
  // B-fragment coords (16-bit B 32x16): N = lane&15, K run = 16*(lane>=16)
  const int bn  = lane & 15;
  const int bKb = (lane >> 4) * 16;
  // C-fragment coords (f32 16x16): N = lane&15, rows = i + 8*(lane>=16)
  const int cn   = lane & 15;
  const int half = lane >> 4;

  // Q fragments held in registers for the whole K/V sweep
  const float* qrow = Qg + (size_t)(qbase + arow) * ND;
  const v16h aq0 = cat16(cvt8(qrow + abase),      cvt8(qrow + abase + 8));
  const v16h aq1 = cat16(cvt8(qrow + 32 + abase), cvt8(qrow + 32 + abase + 8));

  float m[8], l[8];
  v8f acc[4];
#pragma unroll
  for (int i = 0; i < 8; ++i) { m[i] = -INFINITY; l[i] = 0.f; }
#pragma unroll
  for (int d = 0; d < 4; ++d)
#pragma unroll
    for (int i = 0; i < 8; ++i) acc[d][i] = 0.f;

  const float scale = 0.125f;  // 1/sqrt(64)

  // staging coordinates (4 float4 of K and V per thread per tile)
  const int sr  = tid >> 4;          // rows 0..7 (+8 per step)
  const int sc4 = (tid & 15) * 4;    // d-offset
  v4f kreg[4], vreg[4];

  // software pipeline: load tile regs -> (barrier; load next; compute; store next)
  auto load_tile = [&](int kt) {
#pragma unroll
    for (int it = 0; it < 4; ++it) {
      const float* kp = Kg + (size_t)(kt * KTILE + sr + it * 8) * ND + sc4;
      const float* vp = Vg + (size_t)(kt * KTILE + sr + it * 8) * ND + sc4;
      kreg[it] = *(const v4f*)kp;
      vreg[it] = *(const v4f*)vp;
      if (kt + 1 < NKT) {          // prefetch two tiles ahead of compute
        __builtin_prefetch(kp + (size_t)KTILE * ND, 0, 1);
        __builtin_prefetch(vp + (size_t)KTILE * ND, 0, 1);
      }
    }
  };
  auto store_tile = [&](int buf) {
#pragma unroll
    for (int it = 0; it < 4; ++it) {
      const int r = sr + it * 8;
#pragma unroll
      for (int j = 0; j < 4; ++j) {
        Ks[buf][r][sc4 + j] = (_Float16)kreg[it][j];
        Vt[buf][sc4 + j][r] = (_Float16)vreg[it][j];
      }
    }
  };

  load_tile(0);
  store_tile(0);

  for (int kt = 0; kt < NKT; ++kt) {
    __syncthreads();               // tile (kt&1) visible to all waves
    const int buf = kt & 1;
    const int kbase = kt * KTILE;

    if (kt + 1 < NKT) load_tile(kt + 1);   // issue next-tile global loads now

    // ---- gemm1: S = Q * K^T (16 q-rows x 32 k-cols) ----
    v8f sc[2];
#pragma unroll
    for (int kc = 0; kc < 2; ++kc) {
      const _Float16* kr = &Ks[buf][kc * 16 + bn][bKb];
      v16h b0 = cat16(*(const v8h*)kr,        *(const v8h*)(kr + 8));   // d 0..31
      v16h b1 = cat16(*(const v8h*)(kr + 32), *(const v8h*)(kr + 40));  // d 32..63
      v8f c = {};
      c = __builtin_amdgcn_wmma_f32_16x16x32_f16(false, aq0, false, b0,
                                                 (short)0, c, false, false);
      c = __builtin_amdgcn_wmma_f32_16x16x32_f16(false, aq1, false, b1,
                                                 (short)0, c, false, false);
      sc[kc] = c;
    }

    // scale + additive attention mask (L2-resident, shared across heads)
#pragma unroll
    for (int kc = 0; kc < 2; ++kc) {
      const float* mp = Mg + (size_t)(qbase + 8 * half) * NS + (kbase + kc * 16 + cn);
#pragma unroll
      for (int i = 0; i < 8; ++i)
        sc[kc][i] = sc[kc][i] * scale + mp[(size_t)i * NS];
    }

    // ---- online softmax (DPP16 reductions, no LDS round-trips) ----
    float rsc[8];
#pragma unroll
    for (int i = 0; i < 8; ++i) {
      const float t  = red16_max(fmaxf(sc[0][i], sc[1][i]));
      const float mn = fmaxf(m[i], t);
      rsc[i] = __expf(m[i] - mn);
      m[i] = mn;
      l[i] *= rsc[i];
    }
#pragma unroll
    for (int d = 0; d < 4; ++d)
#pragma unroll
      for (int i = 0; i < 8; ++i) acc[d][i] *= rsc[i];

#pragma unroll
    for (int kc = 0; kc < 2; ++kc)
#pragma unroll
      for (int i = 0; i < 8; ++i)
        sc[kc][i] = __expf(sc[kc][i] - m[i]);
#pragma unroll
    for (int i = 0; i < 8; ++i)
      l[i] += red16_sum(sc[0][i] + sc[1][i]);   // denominator is pre-dropout

    // dropout (streamed non-temporal) then pack P to LDS as f16
#pragma unroll
    for (int kc = 0; kc < 2; ++kc) {
      const float* dp = Dg + (size_t)(qbase + 8 * half) * NS + (kbase + kc * 16 + cn);
#pragma unroll
      for (int i = 0; i < 8; ++i) {
        float p = sc[kc][i] * __builtin_nontemporal_load(dp + (size_t)i * NS);
        Ps[wave][i + 8 * half][kc * 16 + cn] = (_Float16)p;
      }
    }

    // ---- gemm2: acc += P * V ----
    const _Float16* pr = &Ps[wave][arow][0];
    v16h ap = cat16(*(const v8h*)(pr + abase), *(const v8h*)(pr + abase + 8));
#pragma unroll
    for (int dc = 0; dc < 4; ++dc) {
      const _Float16* vp2 = &Vt[buf][dc * 16 + bn][bKb];
      v16h bv = cat16(*(const v8h*)vp2, *(const v8h*)(vp2 + 8));
      acc[dc] = __builtin_amdgcn_wmma_f32_16x16x32_f16(false, ap, false, bv,
                                                       (short)0, acc[dc], false, false);
    }

    if (kt + 1 < NKT) store_tile((kt + 1) & 1);  // fill other buffer for next iter
  }

  // epilogue: normalize by softmax sum, store f32 (coalesced across lanes)
#pragma unroll
  for (int dc = 0; dc < 4; ++dc) {
    float* orow = Og + (size_t)(qbase + 8 * half) * ND + dc * 16 + cn;
#pragma unroll
    for (int i = 0; i < 8; ++i)
      orow[(size_t)i * ND] = acc[dc][i] / l[i];
  }
}

extern "C" void kernel_launch(void* const* d_in, const int* in_sizes, int n_in,
                              void* d_out, int out_size, void* d_ws, size_t ws_size,
                              hipStream_t stream) {
  const float* q    = (const float*)d_in[0];
  const float* k    = (const float*)d_in[1];
  const float* v    = (const float*)d_in[2];
  const float* mask = (const float*)d_in[3];
  const float* drop = (const float*)d_in[4];
  float* out = (float*)d_out;

  dim3 grid(NS / QB, NB * NH);   // (32 q-blocks, 32 (b,h) heads)
  dim3 block(128);               // 4 wave32 waves
  hipLaunchKernelGGL(fattn_wmma_f16, grid, block, 0, stream,
                     q, k, v, mask, drop, out);
}
